// BiLSTM_CRF_62070867362014
// MI455X (gfx1250) — matrix operand order repaired and checked by reading, not compiled, not run on hardware
//
#include <hip/hip_runtime.h>
#include <hip/hip_bf16.h>

// ---------------- problem constants ----------------
#define T_LEN 4096
#define EMB   1024
#define HDIM  512          // per-direction hidden
#define G4    2048         // 4*HDIM gate rows
#define KTAG  12
#define START_TAG 10
#define STOP_TAG  11
#define NEG_INF  -10000.0f
#define NCHUNK (EMB / 64)  // K chunks of 64 in the GEMM
#define MTILE  64          // M rows per block (4 x 16 per wave)

typedef __attribute__((ext_vector_type(16))) __bf16       v16bf;
typedef __attribute__((ext_vector_type(8)))  float        v8f;
typedef __attribute__((ext_vector_type(8)))  unsigned int v8u;
typedef __attribute__((ext_vector_type(4)))  unsigned int v4u;

// ---------------- helpers ----------------
__device__ __forceinline__ unsigned short f2bf(float f) {
    unsigned int u = __builtin_bit_cast(unsigned int, f);
    unsigned int r = u + 0x7FFFu + ((u >> 16) & 1u);   // round-to-nearest-even
    return (unsigned short)(r >> 16);
}
__device__ __forceinline__ float bf2f(unsigned short h) {
    unsigned int u = ((unsigned int)h) << 16;
    return __builtin_bit_cast(float, u);
}
__device__ __forceinline__ float sigmoidf_(float x) {
    return 1.0f / (1.0f + __expf(-x));
}
__device__ __forceinline__ unsigned int pack_bf2(float lo, float hi) {
    return (unsigned int)f2bf(lo) | ((unsigned int)f2bf(hi) << 16);
}

// ---------------- Tensor Data Mover: 2-D tile -> LDS (ISA ch.7/8) ----------------
__device__ __forceinline__ void tdm_load_tile_2d(unsigned lds_off, const void* gptr,
                                                 unsigned tile_w, unsigned tile_h,
                                                 unsigned tensor_w, unsigned tensor_h,
                                                 unsigned row_stride_elems) {
    unsigned long long ga = (unsigned long long)gptr;
    v4u g0;
    g0[0] = 1u;                                               // count=1, user mode
    g0[1] = lds_off;                                          // lds_addr (bytes)
    g0[2] = (unsigned int)ga;                                 // global_addr[31:0]
    g0[3] = (unsigned int)((ga >> 32) & 0x01FFFFFFu)          // global_addr[56:32]
          | 0x80000000u;                                      // type=2 ("image")
    v8u g1;
    g1[0] = 1u << 16;                                         // data_size=1 -> 2 bytes
    g1[1] = (tensor_w & 0xFFFFu) << 16;                       // tensor_dim0 lo @ bit48
    g1[2] = (tensor_w >> 16) | ((tensor_h & 0xFFFFu) << 16);  // dim0 hi | dim1 lo
    g1[3] = (tensor_h >> 16) | (tile_w << 16);                // dim1 hi | tile_dim0
    g1[4] = tile_h;                                           // tile_dim1 (tile_dim2=0)
    g1[5] = row_stride_elems;                                 // tensor_dim0_stride lo32
    g1[6] = 0u;
    g1[7] = 0u;
    asm volatile("tensor_load_to_lds %0, %1" :: "s"(g0), "s"(g1) : "memory");
}

// ---------------- prep kernels ----------------
__global__ void __launch_bounds__(256)
gather_embed_kernel(const int* __restrict__ sent, const float* __restrict__ embed,
                    unsigned short* __restrict__ Xbf) {
    long i8 = ((long)blockIdx.x * 256 + threadIdx.x) * 8;
    if (i8 >= (long)T_LEN * EMB) return;
    int t = (int)(i8 >> 10);
    int e = (int)(i8 & (EMB - 1));
    const float* src = embed + (long)sent[t] * EMB + e;
    float4 a = *(const float4*)src;
    float4 b = *(const float4*)(src + 4);
    uint4 p;
    p.x = pack_bf2(a.x, a.y);
    p.y = pack_bf2(a.z, a.w);
    p.z = pack_bf2(b.x, b.y);
    p.w = pack_bf2(b.z, b.w);
    *(uint4*)(Xbf + i8) = p;
}

__global__ void __launch_bounds__(256)
cvt_bf16_kernel(const float* __restrict__ in, unsigned short* __restrict__ out, long n) {
    long i8 = ((long)blockIdx.x * 256 + threadIdx.x) * 8;
    if (i8 >= n) return;
    float4 a = *(const float4*)(in + i8);
    float4 b = *(const float4*)(in + i8 + 4);
    uint4 p;
    p.x = pack_bf2(a.x, a.y);
    p.y = pack_bf2(a.z, a.w);
    p.z = pack_bf2(b.x, b.y);
    p.w = pack_bf2(b.z, b.w);
    *(uint4*)(out + i8) = p;
}

// ---------------- WMMA GEMM: G = X @ w_ih^T + b, both directions ----------------
// grid: (T/64, G4/128, 2), block 256 (8 waves). Block tile 64(M)x128(N);
// each wave holds four 16x16 accumulators along M and reuses one B fragment
// across four A fragments (4 independent back-to-back WMMAs per K-step).
// TDM double-buffers A (64x64, [m][k]) and B (128x64, [n][k]) tiles into LDS.
__global__ void __launch_bounds__(256)
gemm_gates_kernel(const unsigned short* __restrict__ Xbf,
                  const unsigned short* __restrict__ WihF,
                  const unsigned short* __restrict__ WihB,
                  const float* __restrict__ bF, const float* __restrict__ bB,
                  float* __restrict__ Gf, float* __restrict__ Gb) {
    const int dir = blockIdx.z;
    const unsigned short* __restrict__ Wt = dir ? WihB : WihF;  // [G4][EMB] bf16
    const float* __restrict__ bias        = dir ? bB  : bF;
    float* __restrict__ Gout              = dir ? Gb  : Gf;

    const int t0 = blockIdx.x * MTILE;
    const int n0 = blockIdx.y * 128;

    __shared__ unsigned short Als[2][MTILE * 64];  // A: [m][k]
    __shared__ unsigned short Bls[2][128 * 64];    // B: [n][k]

    const int tid  = threadIdx.x;
    const int wid  = tid >> 5;
    const int lane = tid & 31;
    const int half = lane >> 4;
    const int l16  = lane & 15;

    const unsigned short* Abase = Xbf + (long)t0 * EMB;   // + k0
    const unsigned short* Bbase = Wt  + (long)n0 * EMB;   // + k0

    if (wid == 0) {  // prologue: DMA chunk 0 into buffer 0
        tdm_load_tile_2d((unsigned)(unsigned long long)(void*)&Als[0][0], Abase,
                         64, MTILE, EMB, T_LEN, EMB);
        tdm_load_tile_2d((unsigned)(unsigned long long)(void*)&Bls[0][0], Bbase,
                         64, 128, EMB, G4, EMB);
        __builtin_amdgcn_s_wait_tensorcnt(0);
    }
    __syncthreads();

    v8f c0a = {}, c1a = {}, c2a = {}, c3a = {};
    for (int kc = 0; kc < NCHUNK; ++kc) {
        const int buf = kc & 1;
        if (wid == 0 && kc + 1 < NCHUNK) {       // DMA next chunk into other buffer
            const int k1 = (kc + 1) * 64;
            tdm_load_tile_2d((unsigned)(unsigned long long)(void*)&Als[buf ^ 1][0],
                             Abase + k1, 64, MTILE, EMB, T_LEN, EMB);
            tdm_load_tile_2d((unsigned)(unsigned long long)(void*)&Bls[buf ^ 1][0],
                             Bbase + k1, 64, 128, EMB, G4, EMB);
        }
        if (wid == 1 && kc + 2 < NCHUNK)         // warm L2 ahead of the TDM
            __builtin_prefetch(Bbase + (long)(lane * 4) * EMB + (kc + 2) * 64, 0, 0);

        #pragma unroll
        for (int kk = 0; kk < 64; kk += 32) {
            // B fragment: N = wid*16 + l16; element e: K = kk + 8*half + (e<8 ? e : 8+e)
            v8u bu;
            const unsigned int* brow =
                (const unsigned int*)(&Bls[buf][0] + (wid * 16 + l16) * 64);
            #pragma unroll
            for (int v = 0; v < 8; ++v) bu[v] = brow[(kk + 8 * half + 2 * (v & 3) + 16 * (v >> 2)) >> 1];
            v16bf b = __builtin_bit_cast(v16bf, bu);

            // 4 A fragments along M, reusing b (independent accumulators)
            #pragma unroll
            for (int ms = 0; ms < 4; ++ms) {
                v8u au;
                const unsigned int* arow =
                    (const unsigned int*)(&Als[buf][0] + (ms * 16 + l16) * 64);
                #pragma unroll
                for (int v = 0; v < 4; ++v) au[v]     = arow[(kk + 2 * v + 8 * half) >> 1];
                #pragma unroll
                for (int v = 0; v < 4; ++v) au[4 + v] = arow[(kk + 16 + 2 * v + 8 * half) >> 1];
                v16bf a = __builtin_bit_cast(v16bf, au);
                v8f& acc = (ms == 0) ? c0a : (ms == 1) ? c1a : (ms == 2) ? c2a : c3a;
                acc = __builtin_amdgcn_wmma_f32_16x16x32_bf16(false, a, false, b,
                                                              (short)0, acc, false, false);
            }
        }

        if (wid == 0)
            __builtin_amdgcn_s_wait_tensorcnt(0);  // next buffer landed
        __syncthreads();                            // publish to all waves
    }

    const int j  = n0 + wid * 16 + l16;
    const float bv = bias[j];
    #pragma unroll
    for (int ms = 0; ms < 4; ++ms) {
        v8f& acc = (ms == 0) ? c0a : (ms == 1) ? c1a : (ms == 2) ? c2a : c3a;
        #pragma unroll
        for (int v = 0; v < 8; ++v) {
            int m = ms * 16 + v + 8 * half;
            Gout[(long)(t0 + m) * G4 + j] = acc[v] + bv;
        }
    }
}

// ---------------- LSTM recurrence ----------------
// 32 persistent blocks: [0..15] forward, [16..31] backward. Each block keeps a
// 128x512 bf16 w_hh slice (4 gates x 32 channels) LDS-resident, owns its c-slice,
// and exchanges only the 512-float h vector per step through L2.
__global__ void __launch_bounds__(256)
lstm_rec_kernel(const float* __restrict__ Gf, const float* __restrict__ Gb,
                const unsigned short* __restrict__ WhhF,
                const unsigned short* __restrict__ WhhB,
                const float* __restrict__ h0, const float* __restrict__ c0,
                float* __restrict__ hf, float* __restrict__ hb,
                unsigned int* __restrict__ ctr) {
    extern __shared__ char smem[];
    unsigned short* Wl    = (unsigned short*)smem;                 // 128*512*2 B
    float*          hprev = (float*)(smem + 128 * 512 * 2);        // 512
    float*          red   = hprev + 512;                           // 256
    float*          gv    = red + 256;                             // 128
    float*          cl    = gv + 128;                              // 32

    const int tid   = threadIdx.x;
    const int dir   = blockIdx.x >> 4;
    const int blk   = blockIdx.x & 15;
    const int hbase = blk * 32;

    const float* __restrict__ G            = dir ? Gb   : Gf;
    const unsigned short* __restrict__ Whh = dir ? WhhB : WhhF;
    float* __restrict__ hout               = dir ? hb   : hf;
    unsigned int* myctr = ctr + dir;

    for (int i = tid; i < 128 * 512; i += 256) {
        int lr = i >> 9, h = i & 511;
        int grow = (lr >> 5) * HDIM + hbase + (lr & 31);
        Wl[i] = Whh[(long)grow * HDIM + h];
    }
    if (tid < 32) cl[tid] = c0[dir * HDIM + hbase + tid];
    __syncthreads();

    for (int s = 0; s < T_LEN; ++s) {
        const int t = dir ? (T_LEN - 1 - s) : s;
        if (s == 0) {
            for (int i = tid; i < HDIM; i += 256) hprev[i] = h0[dir * HDIM + i];
        } else {
            const int tp = dir ? (t + 1) : (t - 1);
            for (int i = tid; i < HDIM; i += 256) hprev[i] = hout[(long)tp * HDIM + i];
        }
        __syncthreads();

        const int r = tid >> 1, seg = tid & 1;
        const unsigned int* wrow = (const unsigned int*)(Wl + r * 512) + seg * 128;
        const float* hseg = hprev + seg * 256;
        float acc = 0.0f;
        #pragma unroll 4
        for (int k = 0; k < 128; ++k) {
            unsigned int w = wrow[k];
            acc += bf2f((unsigned short)(w & 0xFFFFu)) * hseg[2 * k];
            acc += bf2f((unsigned short)(w >> 16))     * hseg[2 * k + 1];
        }
        red[tid] = acc;
        __syncthreads();

        if (tid < 128) {
            int grow = (tid >> 5) * HDIM + hbase + (tid & 31);
            gv[tid] = red[2 * tid] + red[2 * tid + 1] + G[(long)t * G4 + grow];
        }
        __syncthreads();

        if (tid < 32) {
            float iv = sigmoidf_(gv[tid]);
            float fv = sigmoidf_(gv[32 + tid]);
            float gg = tanhf(gv[64 + tid]);
            float ov = sigmoidf_(gv[96 + tid]);
            float cc = fv * cl[tid] + iv * gg;
            cl[tid] = cc;
            hout[(long)t * HDIM + hbase + tid] = ov * tanhf(cc);
        }

        __threadfence();
        __syncthreads();
        if (tid == 0) {
            __hip_atomic_fetch_add(myctr, 1u, __ATOMIC_ACQ_REL, __HIP_MEMORY_SCOPE_AGENT);
            const unsigned int target = 16u * (unsigned int)(s + 1);
            while (__hip_atomic_load(myctr, __ATOMIC_ACQUIRE, __HIP_MEMORY_SCOPE_AGENT) < target)
                __builtin_amdgcn_s_sleep(2);
        }
        __builtin_amdgcn_s_cluster_barrier();  // NOP when unclustered
        __syncthreads();
    }
}

// ---------------- feats = concat(hf,hb) @ W_out^T + b_out ----------------
__global__ void __launch_bounds__(256)
feats_kernel(const float* __restrict__ hf, const float* __restrict__ hb,
             const float* __restrict__ W_out, const float* __restrict__ b_out,
             float* __restrict__ feats) {
    __shared__ float hcat[1024];
    __shared__ float red[192];
    const int t = blockIdx.x, tid = threadIdx.x;
    for (int i = tid; i < HDIM; i += 256) {
        hcat[i]        = hf[(long)t * HDIM + i];
        hcat[HDIM + i] = hb[(long)t * HDIM + i];
    }
    __syncthreads();
    if (tid < 192) {
        int k = tid >> 4, seg = tid & 15;
        const float* w = W_out + (long)k * 1024 + seg * 64;
        const float* h = hcat + seg * 64;
        float acc = 0.0f;
        #pragma unroll 4
        for (int j = 0; j < 64; ++j) acc += w[j] * h[j];
        red[tid] = acc;
    }
    __syncthreads();
    if (tid < KTAG) {
        float s = b_out[tid];
        #pragma unroll
        for (int u = 0; u < 16; ++u) s += red[tid * 16 + u];
        feats[(long)t * KTAG + tid] = s;
    }
}

// ---------------- Viterbi (single wave, sequential lattice + backtrack) ----------------
__global__ void __launch_bounds__(32)
viterbi_kernel(const float* __restrict__ feats, const float* __restrict__ trans,
               int* __restrict__ bp, float* __restrict__ out) {
    __shared__ float tl[KTAG * KTAG];
    __shared__ float fv[KTAG];
    __shared__ float nfv[KTAG];
    const int tid = threadIdx.x;
    for (int i = tid; i < KTAG * KTAG; i += 32) tl[i] = trans[i];
    if (tid < KTAG) fv[tid] = (tid == START_TAG) ? 0.0f : NEG_INF;
    __syncthreads();

    for (int t = 0; t < T_LEN; ++t) {
        if (tid < KTAG) {
            float best = -3.4e38f; int arg = 0;
            #pragma unroll
            for (int j = 0; j < KTAG; ++j) {
                float v = fv[j] + tl[tid * KTAG + j];
                if (v > best) { best = v; arg = j; }
            }
            bp[t * KTAG + tid] = arg;
            nfv[tid] = best + feats[(long)t * KTAG + tid];
        }
        __syncthreads();
        if (tid < KTAG) fv[tid] = nfv[tid];
        __syncthreads();
    }

    if (tid == 0) {
        float best = -3.4e38f; int arg = 0;
        for (int i = 0; i < KTAG; ++i) {
            float v = fv[i] + tl[STOP_TAG * KTAG + i];
            if (v > best) { best = v; arg = i; }
        }
        out[0] = best;
        int tag = arg;
        for (int t = T_LEN - 1; t >= 0; --t) {
            out[1 + t] = (float)tag;
            tag = bp[t * KTAG + tag];
        }
    }
}

// ---------------- launch ----------------
extern "C" void kernel_launch(void* const* d_in, const int* in_sizes, int n_in,
                              void* d_out, int out_size, void* d_ws, size_t ws_size,
                              hipStream_t stream) {
    const int*   sentence = (const int*)d_in[0];
    const float* embed    = (const float*)d_in[1];
    const float* w_ih_f   = (const float*)d_in[2];
    const float* w_hh_f   = (const float*)d_in[3];
    const float* b_f      = (const float*)d_in[4];
    const float* w_ih_b   = (const float*)d_in[5];
    const float* w_hh_b   = (const float*)d_in[6];
    const float* b_b      = (const float*)d_in[7];
    const float* h0       = (const float*)d_in[8];
    const float* c0       = (const float*)d_in[9];
    const float* W_out    = (const float*)d_in[10];
    const float* b_out    = (const float*)d_in[11];
    const float* trans    = (const float*)d_in[12];

    char* ws = (char*)d_ws;
    size_t off = 0;
    auto alloc = [&](size_t bytes) -> void* {
        void* p = ws + off;
        off = (off + bytes + 255) & ~(size_t)255;
        return p;
    };
    unsigned short* Xbf   = (unsigned short*)alloc((size_t)T_LEN * EMB * 2);
    unsigned short* WihFb = (unsigned short*)alloc((size_t)G4 * EMB * 2);
    unsigned short* WihBb = (unsigned short*)alloc((size_t)G4 * EMB * 2);
    unsigned short* WhhFb = (unsigned short*)alloc((size_t)G4 * HDIM * 2);
    unsigned short* WhhBb = (unsigned short*)alloc((size_t)G4 * HDIM * 2);
    float*          Gf    = (float*)alloc((size_t)T_LEN * G4 * 4);
    float*          Gb    = (float*)alloc((size_t)T_LEN * G4 * 4);
    float*          hf    = (float*)alloc((size_t)T_LEN * HDIM * 4);
    float*          hb    = (float*)alloc((size_t)T_LEN * HDIM * 4);
    float*          feats = (float*)alloc((size_t)T_LEN * KTAG * 4);
    int*            bp    = (int*)alloc((size_t)T_LEN * KTAG * 4);
    unsigned int*   ctr   = (unsigned int*)alloc(256);

    hipMemsetAsync(ctr, 0, 2 * sizeof(unsigned int), stream);

    gather_embed_kernel<<<(T_LEN * EMB / 8) / 256, 256, 0, stream>>>(sentence, embed, Xbf);
    cvt_bf16_kernel<<<(G4 * EMB / 8) / 256, 256, 0, stream>>>(w_ih_f, WihFb, (long)G4 * EMB);
    cvt_bf16_kernel<<<(G4 * EMB / 8) / 256, 256, 0, stream>>>(w_ih_b, WihBb, (long)G4 * EMB);
    cvt_bf16_kernel<<<(G4 * HDIM / 8) / 256, 256, 0, stream>>>(w_hh_f, WhhFb, (long)G4 * HDIM);
    cvt_bf16_kernel<<<(G4 * HDIM / 8) / 256, 256, 0, stream>>>(w_hh_b, WhhBb, (long)G4 * HDIM);

    // TDM-fed WMMA GEMM for gate pre-activations, both directions (64x128 block tiles)
    gemm_gates_kernel<<<dim3(T_LEN / MTILE, G4 / 128, 2), 256, 0, stream>>>(
        Xbf, WihFb, WihBb, b_f, b_b, Gf, Gb);

    const size_t rec_smem = (size_t)128 * 512 * 2 + (512 + 256 + 128 + 32) * sizeof(float);
    lstm_rec_kernel<<<32, 256, rec_smem, stream>>>(Gf, Gb, WhhFb, WhhBb, h0, c0, hf, hb, ctr);

    feats_kernel<<<T_LEN, 256, 0, stream>>>(hf, hb, W_out, b_out, feats);

    viterbi_kernel<<<1, 32, 0, stream>>>(feats, trans, bp, (float*)d_out);
}
